// SimpleMLGKAClassifier_16415365005848
// MI455X (gfx1250) — matrix operand order repaired and checked by reading, not compiled, not run on hardware
//
#include <hip/hip_runtime.h>

// ---------------------------------------------------------------------------
// SimpleMLGKAClassifier on MI455X (gfx1250, wave32, WMMA + async-to-LDS).
//
// Shapes: VOCAB=32000 D=512 H=8 HK=2 DH=64 M=64 DFF=2048 NCLS=2 G=4
//         B=8 S=8192  -> T = 65536 tokens.
//
// All dense GEMMs: bf16 v_wmma_f32_16x16x32_bf16 with f32 accum.
// Weights are pre-transposed to [N][K] bf16 so both A and B LDS tiles are
// filled by straight GLOBAL_LOAD_ASYNC_TO_LDS_B128 copies (ASYNCcnt),
// double-buffered, and every fragment load is 2x ds_load_b128.
// ---------------------------------------------------------------------------

typedef unsigned short u16;
typedef unsigned int u32;
typedef __bf16 bf16x16 __attribute__((ext_vector_type(16)));
typedef float floatx8 __attribute__((ext_vector_type(8)));

#define TOK   65536      // B*S
#define BD    512        // model dim D
#define SEQ   8192
#define BATCH 8
#define NHK   2
#define NH    8
#define GRP   4
#define DHEAD 64
#define FEATM 64
#define DFFN  2048
#define EPSV  1e-6f

// ---------------- scalar helpers ----------------
__device__ __forceinline__ u16 f2bf(float f) {
  u32 u = __float_as_uint(f);
  u += 0x7FFFu + ((u >> 16) & 1u);   // round-to-nearest-even
  return (u16)(u >> 16);
}
__device__ __forceinline__ float bf2f(u16 h) {
  return __uint_as_float(((u32)h) << 16);
}

// ---------------- CDNA5 async copy (global -> LDS, 16B per lane) ----------
// Generic LDS pointer truncated to 32 bits == LDS byte offset (ISA 10.2).
__device__ __forceinline__ void async_b128(void* lds, const void* g) {
  unsigned l = (unsigned)(size_t)lds;
  unsigned long long a = (unsigned long long)(size_t)g;
  asm volatile("global_load_async_to_lds_b128 %0, %1, off"
               :: "v"(l), "v"(a) : "memory");
}
__device__ __forceinline__ void wait_async0() {
  asm volatile("s_wait_asynccnt 0x0" ::: "memory");
}

// ---------------- WMMA fragment loaders (contiguous k = 16*khalf + s, same order for A and B) ----------------
// A 16x32 bf16 (MxK): lane L(0-15)=row M, khalf=L>>4.
//   dword v: K-pair index = khalf*8 + v  -> 2x ds_load_b128.
__device__ __forceinline__ bf16x16 frag_a(const u32* rowbase, int khalf) {
  union { bf16x16 v; u32 u[8]; } f;
#pragma unroll
  for (int e = 0; e < 8; ++e)
    f.u[e] = rowbase[(khalf << 3) + e];
  return f.v;
}
// B 32x16 bf16 (KxN): lane L(0-15)=col N holding K=0..15 sequentially,
// lanes 16-31 hold K=16..31. LDS B tile is n-major [64][32] (from W^T), so
// a fragment is 8 consecutive dwords per lane -> 2x ds_load_b128.
__device__ __forceinline__ bf16x16 frag_b(const u32* colbase, int khalf) {
  union { bf16x16 v; u32 u[8]; } f;
#pragma unroll
  for (int e = 0; e < 8; ++e)
    f.u[e] = colbase[(khalf << 3) + e];
  return f.v;
}

// ---------------- generic WMMA GEMM ----------------
// C[M x N] = A[M x K](bf16,row-major,lda,col off acol) * B^T[N x K](bf16,ldbK)
// Block tile 128x64, BK=32, 256 threads = 8 waves, each wave 32x32.
// Double-buffered LDS, async global->LDS staging.
// EPI: 0 = store bf16 ; 1 = store bf16 exp(-|x|) ; 2 = fp32 C += x (residual)
template <int EPI>
__global__ __launch_bounds__(256) void gemm_wmma(
    const u16* __restrict__ A, const u16* __restrict__ Bt, void* __restrict__ Cv,
    int K, int lda, int ldbK, int ldc, int acol, int ccol) {
  __shared__ __align__(16) u16 As[2][128 * 32];
  __shared__ __align__(16) u16 Bs[2][64 * 32];   // n-major: Bs[n][k]

  const int tid = threadIdx.x, lane = tid & 31, wv = tid >> 5;
  const int waveM = wv >> 1, waveN = wv & 1;
  const int bm = blockIdx.x * 128, bn = blockIdx.y * 64;

  // per-thread staging slices (uniform across the block, no divergence)
  const int ar = tid >> 1, akk = (tid & 1) << 4;   // A: 32B of one row-half
  const int bnr = tid >> 2, bkk = (tid & 3) << 3;  // B: 16B of one n-row
  const u16* gA = A + (size_t)(bm + ar) * lda + acol + akk;
  const u16* gB = Bt + (size_t)(bn + bnr) * ldbK + bkk;

  floatx8 acc[2][2];
#pragma unroll
  for (int i = 0; i < 2; ++i)
#pragma unroll
    for (int j = 0; j < 2; ++j)
#pragma unroll
      for (int r = 0; r < 8; ++r) acc[i][j][r] = 0.f;

  auto issue = [&](int k0, int buf) {
    const u16* pa = gA + k0;
    async_b128(&As[buf][ar * 32 + akk], pa);
    async_b128(&As[buf][ar * 32 + akk + 8], pa + 8);
    async_b128(&Bs[buf][bnr * 32 + bkk], gB + k0);
  };

  issue(0, 0);
  const int nk = K >> 5;
  for (int it = 0; it < nk; ++it) {
    wait_async0();        // my tile-it async writes have landed
    __syncthreads();      // everyone's tile-it landed; prev compute done
    if (it + 1 < nk) issue((it + 1) << 5, (it + 1) & 1);

    const int cur = it & 1;
    const u32* AsU = reinterpret_cast<const u32*>(As[cur]);
    const u32* BsU = reinterpret_cast<const u32*>(Bs[cur]);
    const int lr = lane & 15, khalf = lane >> 4;

    bf16x16 af[2], bfb[2];
#pragma unroll
    for (int s = 0; s < 2; ++s) {
      af[s]  = frag_a(AsU + (size_t)(waveM * 32 + s * 16 + lr) * 16, khalf);
      bfb[s] = frag_b(BsU + (size_t)(waveN * 32 + s * 16 + lr) * 16, khalf);
    }
#pragma unroll
    for (int i = 0; i < 2; ++i)
#pragma unroll
      for (int j = 0; j < 2; ++j)
        acc[i][j] = __builtin_amdgcn_wmma_f32_16x16x32_bf16(
            false, af[i], false, bfb[j], (short)0, acc[i][j], false, false);
  }

  // epilogue: C/D layout -> lane: n = lane&15, m = r + 8*(lane>>4)
  const int lr = lane & 15, lh = lane >> 4;
#pragma unroll
  for (int i = 0; i < 2; ++i)
#pragma unroll
    for (int j = 0; j < 2; ++j) {
      const int rowb = bm + waveM * 32 + i * 16 + lh * 8;
      const int col  = bn + waveN * 32 + j * 16 + lr + ccol;
#pragma unroll
      for (int r = 0; r < 8; ++r) {
        const size_t idx = (size_t)(rowb + r) * ldc + col;
        const float v = acc[i][j][r];
        if (EPI == 0) {
          reinterpret_cast<u16*>(Cv)[idx] = f2bf(v);
        } else if (EPI == 1) {
          reinterpret_cast<u16*>(Cv)[idx] = f2bf(expf(-fabsf(v)));
        } else {
          reinterpret_cast<float*>(Cv)[idx] += v;
        }
      }
    }
}

// ---------------- fused gate/up GEMM: gu = silu(A@Wg) * (A@Wu) ----------------
__global__ __launch_bounds__(256) void gemm_gu_wmma(
    const u16* __restrict__ A, const u16* __restrict__ Bg,
    const u16* __restrict__ Bu, u16* __restrict__ Cgu,
    int K, int lda, int ldbK, int ldc) {
  __shared__ __align__(16) u16 As[2][128 * 32];
  __shared__ __align__(16) u16 Bsg[2][64 * 32];
  __shared__ __align__(16) u16 Bsu[2][64 * 32];

  const int tid = threadIdx.x, lane = tid & 31, wv = tid >> 5;
  const int waveM = wv >> 1, waveN = wv & 1;
  const int bm = blockIdx.x * 128, bn = blockIdx.y * 64;

  const int ar = tid >> 1, akk = (tid & 1) << 4;
  const int bnr = tid >> 2, bkk = (tid & 3) << 3;
  const u16* gA = A + (size_t)(bm + ar) * lda + akk;
  const u16* gBg = Bg + (size_t)(bn + bnr) * ldbK + bkk;
  const u16* gBu = Bu + (size_t)(bn + bnr) * ldbK + bkk;

  floatx8 ag[2][2], au[2][2];
#pragma unroll
  for (int i = 0; i < 2; ++i)
#pragma unroll
    for (int j = 0; j < 2; ++j)
#pragma unroll
      for (int r = 0; r < 8; ++r) { ag[i][j][r] = 0.f; au[i][j][r] = 0.f; }

  auto issue = [&](int k0, int buf) {
    const u16* pa = gA + k0;
    async_b128(&As[buf][ar * 32 + akk], pa);
    async_b128(&As[buf][ar * 32 + akk + 8], pa + 8);
    async_b128(&Bsg[buf][bnr * 32 + bkk], gBg + k0);
    async_b128(&Bsu[buf][bnr * 32 + bkk], gBu + k0);
  };

  issue(0, 0);
  const int nk = K >> 5;
  for (int it = 0; it < nk; ++it) {
    wait_async0();
    __syncthreads();
    if (it + 1 < nk) issue((it + 1) << 5, (it + 1) & 1);

    const int cur = it & 1;
    const u32* AsU = reinterpret_cast<const u32*>(As[cur]);
    const u32* BgU = reinterpret_cast<const u32*>(Bsg[cur]);
    const u32* BuU = reinterpret_cast<const u32*>(Bsu[cur]);
    const int lr = lane & 15, khalf = lane >> 4;

    bf16x16 af[2], bg[2], bu[2];
#pragma unroll
    for (int s = 0; s < 2; ++s) {
      af[s] = frag_a(AsU + (size_t)(waveM * 32 + s * 16 + lr) * 16, khalf);
      bg[s] = frag_b(BgU + (size_t)(waveN * 32 + s * 16 + lr) * 16, khalf);
      bu[s] = frag_b(BuU + (size_t)(waveN * 32 + s * 16 + lr) * 16, khalf);
    }
#pragma unroll
    for (int i = 0; i < 2; ++i)
#pragma unroll
      for (int j = 0; j < 2; ++j) {
        ag[i][j] = __builtin_amdgcn_wmma_f32_16x16x32_bf16(
            false, af[i], false, bg[j], (short)0, ag[i][j], false, false);
        au[i][j] = __builtin_amdgcn_wmma_f32_16x16x32_bf16(
            false, af[i], false, bu[j], (short)0, au[i][j], false, false);
      }
  }

  const int lr = lane & 15, lh = lane >> 4;
#pragma unroll
  for (int i = 0; i < 2; ++i)
#pragma unroll
    for (int j = 0; j < 2; ++j) {
      const int rowb = bm + waveM * 32 + i * 16 + lh * 8;
      const int col  = bn + waveN * 32 + j * 16 + lr;
#pragma unroll
      for (int r = 0; r < 8; ++r) {
        const float g = ag[i][j][r];
        const float u = au[i][j][r];
        const float s = g / (1.f + expf(-g));   // silu
        Cgu[(size_t)(rowb + r) * ldc + col] = f2bf(s * u);
      }
    }
}

// ---------------- embedding gather + RMSNorm ----------------
__global__ __launch_bounds__(256) void embed_rms(
    const int* __restrict__ ids, const float* __restrict__ emb,
    const float* __restrict__ g1, float* __restrict__ xres,
    u16* __restrict__ h) {
  const int t = blockIdx.x, tid = threadIdx.x;
  const float* row = emb + (size_t)ids[t] * BD;
  const float x0 = row[tid], x1 = row[tid + 256];
  __shared__ float red[256];
  red[tid] = x0 * x0 + x1 * x1;
  __syncthreads();
  for (int s = 128; s > 0; s >>= 1) {
    if (tid < s) red[tid] += red[tid + s];
    __syncthreads();
  }
  const float inv = rsqrtf(red[0] * (1.f / BD) + EPSV);
  const size_t base = (size_t)t * BD;
  xres[base + tid]       = x0;
  xres[base + tid + 256] = x1;
  h[base + tid]       = f2bf(x0 * inv * g1[tid]);
  h[base + tid + 256] = f2bf(x1 * inv * g1[tid + 256]);
}

// ---------------- RMSNorm over residual stream ----------------
__global__ __launch_bounds__(256) void rms2(
    const float* __restrict__ xres, const float* __restrict__ g2,
    u16* __restrict__ h) {
  const int t = blockIdx.x, tid = threadIdx.x;
  const size_t base = (size_t)t * BD;
  const float x0 = xres[base + tid], x1 = xres[base + tid + 256];
  __shared__ float red[256];
  red[tid] = x0 * x0 + x1 * x1;
  __syncthreads();
  for (int s = 128; s > 0; s >>= 1) {
    if (tid < s) red[tid] += red[tid + s];
    __syncthreads();
  }
  const float inv = rsqrtf(red[0] * (1.f / BD) + EPSV);
  h[base + tid]       = f2bf(x0 * inv * g2[tid]);
  h[base + tid + 256] = f2bf(x1 * inv * g2[tid + 256]);
}

// ---------------- kv state + z accumulation ----------------
// kv[b,kh,m,d] = sum_s pk[b,s,kh,m]*v[b,s,kh,d] ; z[b,kh,m] = sum_s pk
// grid (B*HK, 16 s-chunks of 512), 256 thr: m = tid&63, dgroup = tid>>6 (16 d).
__global__ __launch_bounds__(256) void kvz_accum(
    const u16* __restrict__ pk, const u16* __restrict__ v,
    float* __restrict__ kv, float* __restrict__ z) {
  const int bk = blockIdx.x, b = bk >> 1, kh = bk & 1;
  const int m = threadIdx.x & 63, dg = threadIdx.x >> 6;
  float acc[16];
#pragma unroll
  for (int j = 0; j < 16; ++j) acc[j] = 0.f;
  float zacc = 0.f;
  const int s0 = blockIdx.y * 512;
  for (int si = 0; si < 512; ++si) {
    const size_t t = (size_t)b * SEQ + s0 + si;
    const u16* pr = pk + t * (NHK * FEATM) + kh * FEATM;
    const u16* vr = v  + t * (NHK * DHEAD) + kh * DHEAD + dg * 16;
    const float pm = bf2f(pr[m]);
    zacc += pm;
#pragma unroll
    for (int j = 0; j < 16; ++j) acc[j] += pm * bf2f(vr[j]);
  }
  float* dst = kv + ((size_t)bk * FEATM + m) * DHEAD + dg * 16;
#pragma unroll
  for (int j = 0; j < 16; ++j) atomicAdd(&dst[j], acc[j]);
  if (dg == 0) atomicAdd(&z[bk * FEATM + m], zacc);
}

// ---------------- attention combine ----------------
// attn[t, h*DH+d] = (sum_m pq*kv) / (sum_m pq*z + eps), h = kh*G+g.
// grid (B*HK, S/64); thread = (token_local = tid>>2, g = tid&3).
__global__ __launch_bounds__(256) void attn_combine(
    const u16* __restrict__ pq, const float* __restrict__ kv,
    const float* __restrict__ z, u16* __restrict__ attn) {
  const int bk = blockIdx.x, b = bk >> 1, kh = bk & 1;
  __shared__ float kvs[FEATM * DHEAD];
  __shared__ float zs[FEATM];
  const float* src = kv + (size_t)bk * FEATM * DHEAD;
  for (int i = threadIdx.x; i < FEATM * DHEAD; i += 256) kvs[i] = src[i];
  if (threadIdx.x < FEATM) zs[threadIdx.x] = z[bk * FEATM + threadIdx.x];
  __syncthreads();

  const int tl = threadIdx.x >> 2, g = threadIdx.x & 3;
  const size_t t = (size_t)b * SEQ + blockIdx.y * 64 + tl;
  const int h = kh * GRP + g;
  const u16* prow = pq + t * (NH * FEATM) + h * FEATM;

  float num[DHEAD];
#pragma unroll
  for (int d = 0; d < DHEAD; ++d) num[d] = 0.f;
  float den = 0.f;
  for (int m = 0; m < FEATM; ++m) {
    const float pm = bf2f(prow[m]);
    den += pm * zs[m];
    const float* kr = kvs + m * DHEAD;
#pragma unroll
    for (int d = 0; d < DHEAD; ++d) num[d] += pm * kr[d];
  }
  const float inv = 1.f / (den + EPSV);
  u16* orow = attn + t * (NH * DHEAD) + h * DHEAD;
#pragma unroll
  for (int d = 0; d < DHEAD; ++d) orow[d] = f2bf(num[d] * inv);
}

// ---------------- mean pool, head, utilities ----------------
__global__ __launch_bounds__(128) void pool_kernel(
    const float* __restrict__ xres, float* __restrict__ pooled) {
  const int b = blockIdx.x, d = blockIdx.y * 128 + threadIdx.x;
  float s = 0.f;
  for (int si = 0; si < SEQ; ++si)
    s += xres[((size_t)b * SEQ + si) * BD + d];
  pooled[b * BD + d] = s * (1.f / SEQ);
}

__global__ void head_kernel(const float* __restrict__ pooled,
                            const float* __restrict__ Wc,
                            const float* __restrict__ bc,
                            float* __restrict__ out) {
  const int i = threadIdx.x;
  if (i >= BATCH * 2) return;
  const int b = i >> 1, c = i & 1;
  float s = bc[c];
  for (int d = 0; d < BD; ++d) s += pooled[b * BD + d] * Wc[d * 2 + c];
  out[b * 2 + c] = s;
}

// transpose + convert: dst[N x K] bf16 = src[K x N]^T fp32 (K,N mult of 32)
__global__ __launch_bounds__(256) void cvt_f2bf_T(
    const float* __restrict__ src, u16* __restrict__ dst, int K, int N) {
  __shared__ u16 tile[32][33];
  const int k0 = blockIdx.x * 32, n0 = blockIdx.y * 32;
  const int tx = threadIdx.x & 31, ty = threadIdx.x >> 5;
#pragma unroll
  for (int r = ty; r < 32; r += 8)
    tile[r][tx] = f2bf(src[(size_t)(k0 + r) * N + n0 + tx]);
  __syncthreads();
#pragma unroll
  for (int r = ty; r < 32; r += 8)
    dst[(size_t)(n0 + r) * K + k0 + tx] = tile[tx][r];
}

__global__ void zero_f32(float* __restrict__ p, int n) {
  const int i = blockIdx.x * 256 + threadIdx.x;
  if (i < n) p[i] = 0.f;
}

// ---------------------------------------------------------------------------
extern "C" void kernel_launch(void* const* d_in, const int* in_sizes, int n_in,
                              void* d_out, int out_size, void* d_ws,
                              size_t ws_size, hipStream_t stream) {
  (void)in_sizes; (void)n_in; (void)out_size;
  const int*   ids  = (const int*)d_in[0];
  const float* emb  = (const float*)d_in[1];
  const float* Wq   = (const float*)d_in[2];
  const float* Wk   = (const float*)d_in[3];
  const float* Wv   = (const float*)d_in[4];
  const float* Wphi = (const float*)d_in[5];
  const float* Wo   = (const float*)d_in[6];
  const float* g1   = (const float*)d_in[7];
  const float* g2   = (const float*)d_in[8];
  const float* Wg   = (const float*)d_in[9];
  const float* Wu   = (const float*)d_in[10];
  const float* Wd   = (const float*)d_in[11];
  const float* Wc   = (const float*)d_in[12];
  const float* bc   = (const float*)d_in[13];

  char* ws = (char*)d_ws;
  size_t off = 0;
  auto alloc = [&](size_t bytes) {
    size_t p = off;
    off = (off + bytes + 255) & ~(size_t)255;
    return p;
  };
  const size_t o_wq   = alloc((size_t)BD * 512 * 2);
  const size_t o_wk   = alloc((size_t)BD * 128 * 2);
  const size_t o_wv   = alloc((size_t)BD * 128 * 2);
  const size_t o_wphi = alloc((size_t)64 * 64 * 2);
  const size_t o_wo   = alloc((size_t)512 * BD * 2);
  const size_t o_wg   = alloc((size_t)BD * DFFN * 2);
  const size_t o_wu   = alloc((size_t)BD * DFFN * 2);
  const size_t o_wd   = alloc((size_t)DFFN * BD * 2);
  const size_t o_x    = alloc((size_t)TOK * BD * 4);
  const size_t o_h    = alloc((size_t)TOK * BD * 2);
  const size_t o_q    = alloc((size_t)TOK * 512 * 2);
  const size_t o_pq   = alloc((size_t)TOK * 512 * 2);
  const size_t o_k    = alloc((size_t)TOK * 128 * 2);
  const size_t o_v    = alloc((size_t)TOK * 128 * 2);
  const size_t o_pk   = alloc((size_t)TOK * 128 * 2);
  const size_t o_attn = alloc((size_t)TOK * 512 * 2);
  const size_t o_kv   = alloc((size_t)BATCH * NHK * FEATM * DHEAD * 4);
  const size_t o_z    = alloc((size_t)BATCH * NHK * FEATM * 4);
  const size_t o_gu   = alloc((size_t)8192 * DFFN * 2);
  const size_t o_pool = alloc((size_t)BATCH * BD * 4);
  if (ws_size < off) return;

  // transposed bf16 weights: stored [N][K]
  u16* wqT   = (u16*)(ws + o_wq);
  u16* wkT   = (u16*)(ws + o_wk);
  u16* wvT   = (u16*)(ws + o_wv);
  u16* wphiT = (u16*)(ws + o_wphi);
  u16* woT   = (u16*)(ws + o_wo);
  u16* wgT   = (u16*)(ws + o_wg);
  u16* wuT   = (u16*)(ws + o_wu);
  u16* wdT   = (u16*)(ws + o_wd);
  float* xres = (float*)(ws + o_x);
  u16* hbuf   = (u16*)(ws + o_h);
  u16* qbuf   = (u16*)(ws + o_q);
  u16* pqbuf  = (u16*)(ws + o_pq);
  u16* kbuf   = (u16*)(ws + o_k);
  u16* vbuf   = (u16*)(ws + o_v);
  u16* pkbuf  = (u16*)(ws + o_pk);
  u16* attnb  = (u16*)(ws + o_attn);
  float* kvb  = (float*)(ws + o_kv);
  float* zb   = (float*)(ws + o_z);
  u16* gubuf  = (u16*)(ws + o_gu);
  float* pool = (float*)(ws + o_pool);

  auto cvtT = [&](const float* s, u16* d, int K, int N) {
    cvt_f2bf_T<<<dim3(K / 32, N / 32), 256, 0, stream>>>(s, d, K, N);
  };
  cvtT(Wq, wqT, BD, 512);
  cvtT(Wk, wkT, BD, 128);
  cvtT(Wv, wvT, BD, 128);
  cvtT(Wphi, wphiT, 64, 64);
  cvtT(Wo, woT, 512, BD);
  cvtT(Wg, wgT, BD, DFFN);
  cvtT(Wu, wuT, BD, DFFN);
  cvtT(Wd, wdT, DFFN, BD);

  // x = emb[ids]; h = rmsnorm(x)*g1
  embed_rms<<<TOK, 256, 0, stream>>>(ids, emb, g1, xres, hbuf);

  // q/k/v projections (bf16 WMMA)
  gemm_wmma<0><<<dim3(TOK / 128, 512 / 64), 256, 0, stream>>>(
      hbuf, wqT, qbuf, BD, BD, BD, 512, 0, 0);
  gemm_wmma<0><<<dim3(TOK / 128, 128 / 64), 256, 0, stream>>>(
      hbuf, wkT, kbuf, BD, BD, BD, 128, 0, 0);
  gemm_wmma<0><<<dim3(TOK / 128, 128 / 64), 256, 0, stream>>>(
      hbuf, wvT, vbuf, BD, BD, BD, 128, 0, 0);

  // Laplacian feature maps: phi = exp(-|q @ Wphi|), per head (fused epilogue)
  for (int h = 0; h < NH; ++h)
    gemm_wmma<1><<<dim3(TOK / 128, 1), 256, 0, stream>>>(
        qbuf, wphiT, pqbuf, DHEAD, 512, DHEAD, 512, h * DHEAD, h * FEATM);
  for (int kh = 0; kh < NHK; ++kh)
    gemm_wmma<1><<<dim3(TOK / 128, 1), 256, 0, stream>>>(
        kbuf, wphiT, pkbuf, DHEAD, 128, DHEAD, 128, kh * DHEAD, kh * FEATM);

  // kv state + normalizer
  {
    const int nfl = BATCH * NHK * FEATM * DHEAD + BATCH * NHK * FEATM;
    zero_f32<<<(nfl + 255) / 256, 256, 0, stream>>>(kvb, nfl);
    kvz_accum<<<dim3(BATCH * NHK, SEQ / 512), 256, 0, stream>>>(pkbuf, vbuf,
                                                                kvb, zb);
  }
  // combine: attn = (pq @ kv) / (pq @ z + eps)
  attn_combine<<<dim3(BATCH * NHK, SEQ / 64), 256, 0, stream>>>(pqbuf, kvb, zb,
                                                                attnb);
  // x += attn @ Wo   (residual epilogue)
  gemm_wmma<2><<<dim3(TOK / 128, 512 / 64), 256, 0, stream>>>(
      attnb, woT, xres, 512, 512, 512, BD, 0, 0);

  // h2 = rmsnorm(x)*g2  (reuse hbuf)
  rms2<<<TOK, 256, 0, stream>>>(xres, g2, hbuf);

  // FFN (chunked over tokens): gu = silu(h2@Wg)*(h2@Wu); x += gu @ Wd
  for (int c = 0; c < TOK / 8192; ++c) {
    const size_t t0 = (size_t)c * 8192;
    gemm_gu_wmma<<<dim3(8192 / 128, DFFN / 64), 256, 0, stream>>>(
        hbuf + t0 * BD, wgT, wuT, gubuf, BD, BD, BD, DFFN);
    gemm_wmma<2><<<dim3(8192 / 128, BD / 64), 256, 0, stream>>>(
        gubuf, wdT, (void*)(xres + t0 * BD), DFFN, DFFN, DFFN, BD, 0, 0);
  }

  // mean pool + linear head
  pool_kernel<<<dim3(BATCH, BD / 128), 128, 0, stream>>>(xres, pool);
  head_kernel<<<1, 32, 0, stream>>>(pool, Wc, bc, (float*)d_out);
}
